// LocalGlobalTransformerEncoderBlock_62191126446773
// MI455X (gfx1250) — compile-verified, matrix operand order checked by reading, not run on hardware
//
#include <hip/hip_runtime.h>
#include <math.h>

#define Bsz 4
#define Seq 2048
#define Dm  512
#define Hn  8
#define HDm 64
#define FFm 2048
#define NTOK (Bsz*Seq)

typedef __bf16 bf16_t;
typedef bf16_t v16bf __attribute__((ext_vector_type(16)));
typedef float  v8f   __attribute__((ext_vector_type(8)));
typedef int    v4i   __attribute__((vector_size(16)));   // matches async-LDS builtin param

struct __align__(16) U128 { unsigned int a, b, c, d; };
union Frag { U128 u[2]; v16bf v; };

#if __has_builtin(__builtin_amdgcn_global_load_async_to_lds_b128) && \
    __has_builtin(__builtin_amdgcn_s_wait_asynccnt)
#define USE_ASYNC_LDS 1
#else
#define USE_ASYNC_LDS 0
#endif

#if USE_ASYNC_LDS
// (global source, LDS destination, imm offset, imm cpol)
#define ASYNC_B128(gsrc, ldst) \
  __builtin_amdgcn_global_load_async_to_lds_b128((v4i*)(gsrc), (v4i*)(ldst), 0, 0)
#endif

__device__ __forceinline__ unsigned short f2bf(float f) {
  unsigned int u = __float_as_uint(f);
  u += 0x7FFFu + ((u >> 16) & 1u);   // round-to-nearest-even
  return (unsigned short)(u >> 16);
}

__device__ __forceinline__ v8f vzero() {
  v8f z;
#pragma unroll
  for (int i = 0; i < 8; ++i) z[i] = 0.0f;
  return z;
}

__device__ __forceinline__ v8f wmma_bf16(const Frag& A, const Frag& B, v8f C) {
  return __builtin_amdgcn_wmma_f32_16x16x32_bf16(false, A.v, false, B.v,
                                                 (short)0, C, false, false);
}

// ---------------- fp32 -> bf16 convert ----------------
__global__ void cvt_bf16_kernel(const float* __restrict__ src,
                                unsigned short* __restrict__ dst, int n) {
  int i = blockIdx.x * blockDim.x + threadIdx.x;
  if (i < n) dst[i] = f2bf(src[i]);
}

// ---------------- LayerNorm (fp32 in, bf16 out), one token per block ----------------
__global__ __launch_bounds__(256)
void ln_bf16_kernel(const float* __restrict__ x, const float* __restrict__ g,
                    const float* __restrict__ bta, unsigned short* __restrict__ out) {
  __shared__ float red[2][8];
  const int row = blockIdx.x;
  const int tid = threadIdx.x;
  const int c0 = tid * 2, c1 = c0 + 1;
  const float v0 = x[(size_t)row * Dm + c0];
  const float v1 = x[(size_t)row * Dm + c1];
  float s = v0 + v1, ss = v0 * v0 + v1 * v1;
#pragma unroll
  for (int mm = 16; mm >= 1; mm >>= 1) {
    s  += __shfl_xor(s,  mm, 32);
    ss += __shfl_xor(ss, mm, 32);
  }
  if ((tid & 31) == 0) { red[0][tid >> 5] = s; red[1][tid >> 5] = ss; }
  __syncthreads();
  s = 0.f; ss = 0.f;
#pragma unroll
  for (int i = 0; i < 8; ++i) { s += red[0][i]; ss += red[1][i]; }
  const float mean = s * (1.0f / Dm);
  const float var  = ss * (1.0f / Dm) - mean * mean;
  const float inv  = rsqrtf(var + 1e-5f);
  out[(size_t)row * Dm + c0] = f2bf((v0 - mean) * inv * g[c0] + bta[c0]);
  out[(size_t)row * Dm + c1] = f2bf((v1 - mean) * inv * g[c1] + bta[c1]);
}

// ---------------- WMMA bf16 GEMM: C[M,N] = A[M,K] * W[N,K]^T + epilogue ----------------
// Block tile 128x128, BK=32, 8 waves; each wave owns a 32x64 tile (2x4 WMMA frags).
// Double-buffered LDS fed by GLOBAL_LOAD_ASYNC_TO_LDS_B128 (ASYNCcnt) when available.
struct GemmParams {
  const unsigned short* A;    // [M,K] bf16
  const unsigned short* W;    // [N,K] bf16 (row-major weight, K contiguous)
  const float* bias;          // [N]
  const float* resid;         // [M,512] for EPI 1
  float* outF;                // fp32 out (EPI 1)
  unsigned short* outH;       // bf16 out (EPI 2, N==FFm)
  unsigned short* qo;         // EPI 0 scatter targets
  unsigned short* ko;
  unsigned short* vto;
  int M, N, K;
};

template <int EPI>
__global__ __launch_bounds__(256, 2)
void gemm_bf16_kernel(GemmParams p) {
  __shared__ __align__(16) unsigned short As[2][128 * 32];
  __shared__ __align__(16) unsigned short Bs[2][128 * 32];
  const int tid  = threadIdx.x;
  const int lane = tid & 31;
  const int wid  = tid >> 5;
  const int wm   = wid >> 1, wn = wid & 1;
  const int lm   = lane & 15, kh = lane >> 4;
  const int n0   = blockIdx.x * 128;
  const int m0   = blockIdx.y * 128;

  // this thread's two 16B staging chunks per matrix per tile
  const int cid0 = tid,       row0 = cid0 >> 2, c80 = (cid0 & 3) << 3;
  const int cid1 = tid + 256, row1 = cid1 >> 2, c81 = (cid1 & 3) << 3;
  const unsigned short* gA0 = p.A + (size_t)(m0 + row0) * p.K + c80;
  const unsigned short* gA1 = p.A + (size_t)(m0 + row1) * p.K + c81;
  const unsigned short* gW0 = p.W + (size_t)(n0 + row0) * p.K + c80;
  const unsigned short* gW1 = p.W + (size_t)(n0 + row1) * p.K + c81;

  v8f acc[2][4];
#pragma unroll
  for (int i = 0; i < 2; ++i)
#pragma unroll
    for (int j = 0; j < 4; ++j) acc[i][j] = vzero();

  const int ksteps = p.K >> 5;

#if USE_ASYNC_LDS
  // prologue: issue tiles 0 and 1 (4 async ops each per wave)
  {
    ASYNC_B128(gA0, &As[0][row0 * 32 + c80]);
    ASYNC_B128(gA1, &As[0][row1 * 32 + c81]);
    ASYNC_B128(gW0, &Bs[0][row0 * 32 + c80]);
    ASYNC_B128(gW1, &Bs[0][row1 * 32 + c81]);
    if (ksteps > 1) {
      ASYNC_B128(gA0 + 32, &As[1][row0 * 32 + c80]);
      ASYNC_B128(gA1 + 32, &As[1][row1 * 32 + c81]);
      ASYNC_B128(gW0 + 32, &Bs[1][row0 * 32 + c80]);
      ASYNC_B128(gW1 + 32, &Bs[1][row1 * 32 + c81]);
    }
  }
#endif

  for (int kt = 0; kt < ksteps; ++kt) {
    const int cur = kt & 1;
#if USE_ASYNC_LDS
    if (kt + 1 < ksteps) __builtin_amdgcn_s_wait_asynccnt(4);  // tile kt landed
    else                 __builtin_amdgcn_s_wait_asynccnt(0);
    __syncthreads();
#else
    const int kbase = kt << 5;
    *(U128*)(&As[cur][row0 * 32 + c80]) = *(const U128*)(gA0 + kbase);
    *(U128*)(&As[cur][row1 * 32 + c81]) = *(const U128*)(gA1 + kbase);
    *(U128*)(&Bs[cur][row0 * 32 + c80]) = *(const U128*)(gW0 + kbase);
    *(U128*)(&Bs[cur][row1 * 32 + c81]) = *(const U128*)(gW1 + kbase);
    if (kt + 1 < ksteps) {
      __builtin_prefetch(gA0 + kbase + 32, 0, 1);
      __builtin_prefetch(gW0 + kbase + 32, 0, 1);
    }
    __syncthreads();
#endif
    // A/B-fragment layout: lane = khalf*16 + row ; 16B chunks at K = khalf*8 and 16+khalf*8
    Frag a[2], b[4];
#pragma unroll
    for (int fm = 0; fm < 2; ++fm) {
      int row = wm * 32 + fm * 16 + lm;
      a[fm].u[0] = *(const U128*)(&As[cur][row * 32 + kh * 8]);
      a[fm].u[1] = *(const U128*)(&As[cur][row * 32 + 16 + kh * 8]);
    }
#pragma unroll
    for (int fn = 0; fn < 4; ++fn) {
      int row = wn * 64 + fn * 16 + lm;
      b[fn].u[0] = *(const U128*)(&Bs[cur][row * 32 + kh * 8]);
      b[fn].u[1] = *(const U128*)(&Bs[cur][row * 32 + 16 + kh * 8]);
    }
#pragma unroll
    for (int fm = 0; fm < 2; ++fm)
#pragma unroll
      for (int fn = 0; fn < 4; ++fn)
        acc[fm][fn] = wmma_bf16(a[fm], b[fn], acc[fm][fn]);
    __syncthreads();
#if USE_ASYNC_LDS
    if (kt + 2 < ksteps) {  // refill the buffer just freed
      const int nb = (kt + 2) << 5;
      ASYNC_B128(gA0 + nb, &As[cur][row0 * 32 + c80]);
      ASYNC_B128(gA1 + nb, &As[cur][row1 * 32 + c81]);
      ASYNC_B128(gW0 + nb, &Bs[cur][row0 * 32 + c80]);
      ASYNC_B128(gW1 + nb, &Bs[cur][row1 * 32 + c81]);
    }
#endif
  }

  // Epilogue: C frag element (r,lane): m = r + 8*(lane/16), n = lane%16
#pragma unroll
  for (int fm = 0; fm < 2; ++fm) {
#pragma unroll
    for (int fn = 0; fn < 4; ++fn) {
#pragma unroll
      for (int r = 0; r < 8; ++r) {
        int m = m0 + wm * 32 + fm * 16 + kh * 8 + r;
        int n = n0 + wn * 64 + fn * 16 + lm;
        float c = acc[fm][fn][r] + p.bias[n];
        if (EPI == 0) {
          // scatter qkv -> q[b,h,s,hd], k[b,h,s,hd], v^T[b,h,hd,s]
          int t = n / Dm, rem = n - t * Dm;
          int h = rem / HDm, hd = rem - h * HDm;
          int bb = m >> 11, s = m & (Seq - 1);
          unsigned short hv = f2bf(c);
          size_t bh = (size_t)bb * Hn + h;
          if (t == 0)      p.qo[(bh * Seq + s) * HDm + hd] = hv;
          else if (t == 1) p.ko[(bh * Seq + s) * HDm + hd] = hv;
          else             p.vto[(bh * HDm + hd) * Seq + s] = hv;
        } else if (EPI == 1) {
          p.outF[(size_t)m * Dm + n] = c + p.resid[(size_t)m * Dm + n];
        } else {
          float gg = 0.5f * c * (1.0f + erff(c * 0.70710678118654752f));
          p.outH[(size_t)m * FFm + n] = f2bf(gg);
        }
      }
    }
  }
}

// ---------------- Local+global attention, flash-style, WMMA ----------------
// 4 waves/block, each wave = one 16-query block of one (b,h).
__global__ __launch_bounds__(128)
void attn_kernel(const unsigned short* __restrict__ qh,
                 const unsigned short* __restrict__ kmat,
                 const unsigned short* __restrict__ vth,
                 const unsigned char*  __restrict__ pad,
                 unsigned short* __restrict__ oh) {
  __shared__ __align__(16) unsigned short Pt[4][16 * 32];
  const int lane = threadIdx.x & 31;
  const int wid  = threadIdx.x >> 5;
  const int qb   = blockIdx.x * 4 + wid;
  const int SB   = Seq / 16;
  const int bh   = qb / SB;
  const int q0   = (qb % SB) * 16;
  const int b    = bh / Hn;
  const int h    = bh % Hn;
  const int lm   = lane & 15, khf = lane >> 4;

  // Q fragments (16 x 64 => two K=32 A-fragments)
  Frag aq[2];
  {
    size_t base = ((size_t)bh * Seq + q0 + lm) * HDm;
#pragma unroll
    for (int w = 0; w < 2; ++w) {
      aq[w].u[0] = *(const U128*)(qh + base + w * 32 + khf * 8);
      aq[w].u[1] = *(const U128*)(qh + base + w * 32 + 16 + khf * 8);
    }
  }

  float rmax[8], rsum[8];
  v8f o[4];
#pragma unroll
  for (int r = 0; r < 8; ++r) { rmax[r] = -3.0e30f; rsum[r] = 0.0f; }
#pragma unroll
  for (int nf = 0; nf < 4; ++nf) o[nf] = vzero();

  // key-chunk segments: local band [q0-64, q0+79] plus chunk 0 (global key);
  // query block 0 contains the global query -> sweep all keys.
  int segs, cs[2], ce[2];
  if (q0 == 0) { segs = 1; cs[0] = 0; ce[0] = Seq / 32 - 1; }
  else {
    int lo = q0 - 64; if (lo < 0) lo = 0;
    int hi = q0 + 79; if (hi > Seq - 1) hi = Seq - 1;
    int c0 = lo >> 5, c1 = hi >> 5;
    if (c0 > 0) { segs = 2; cs[0] = 0; ce[0] = 0; cs[1] = c0; ce[1] = c1; }
    else        { segs = 1; cs[0] = 0; ce[0] = c1; }
  }

  unsigned short* Pw = Pt[wid];

  for (int sg = 0; sg < segs; ++sg) {
    for (int ch = cs[sg]; ch <= ce[sg]; ++ch) {
      const int kb0 = ch * 32;
      // K B-fragments: [key-half][hd-window]
      Frag bk[2][2];
#pragma unroll
      for (int k2 = 0; k2 < 2; ++k2) {
        size_t base = ((size_t)bh * Seq + kb0 + k2 * 16 + lm) * HDm;
#pragma unroll
        for (int w = 0; w < 2; ++w) {
          bk[k2][w].u[0] = *(const U128*)(kmat + base + w * 32 + khf * 8);
          bk[k2][w].u[1] = *(const U128*)(kmat + base + w * 32 + 16 + khf * 8);
        }
      }
      v8f s0 = vzero(), s1 = vzero();
      s0 = wmma_bf16(aq[0], bk[0][0], s0);
      s0 = wmma_bf16(aq[1], bk[0][1], s0);
      s1 = wmma_bf16(aq[0], bk[1][0], s1);
      s1 = wmma_bf16(aq[1], bk[1][1], s1);

      const int kj0 = kb0 + lm, kj1 = kj0 + 16;
      const bool pd0 = pad[b * Seq + kj0] != 0;
      const bool pd1 = pad[b * Seq + kj1] != 0;
      float fac[8];
#pragma unroll
      for (int r = 0; r < 8; ++r) {
        int qi = q0 + r + (khf << 3);
        float v0 = s0[r] * 0.125f;   // HD^-0.5
        float v1 = s1[r] * 0.125f;
        int d0 = qi - kj0; if (d0 < 0) d0 = -d0;
        int d1 = qi - kj1; if (d1 < 0) d1 = -d1;
        bool ok0 = ((d0 <= 64) || (qi == 0) || (kj0 == 0)) && !pd0;
        bool ok1 = ((d1 <= 64) || (qi == 0) || (kj1 == 0)) && !pd1;
        v0 = ok0 ? v0 : -1.0e30f;
        v1 = ok1 ? v1 : -1.0e30f;
        float mloc = fmaxf(v0, v1);
#pragma unroll
        for (int mm = 8; mm >= 1; mm >>= 1)
          mloc = fmaxf(mloc, __shfl_xor(mloc, mm, 16));
        float mnew = fmaxf(rmax[r], mloc);
        float f = __expf(rmax[r] - mnew);
        float p0 = __expf(v0 - mnew);
        float p1 = __expf(v1 - mnew);
        float rs = p0 + p1;
#pragma unroll
        for (int mm = 8; mm >= 1; mm >>= 1)
          rs += __shfl_xor(rs, mm, 16);
        rsum[r] = rsum[r] * f + rs;
        rmax[r] = mnew;
        fac[r] = f;
        int mrow = r + (khf << 3);
        Pw[mrow * 32 + lm]      = f2bf(p0);
        Pw[mrow * 32 + lm + 16] = f2bf(p1);
      }
#pragma unroll
      for (int nf = 0; nf < 4; ++nf)
#pragma unroll
        for (int r = 0; r < 8; ++r) o[nf][r] *= fac[r];

      __builtin_amdgcn_wave_barrier();   // wave-internal LDS ordering (ds ops in-order)
      Frag pa;
      pa.u[0] = *(const U128*)(Pw + lm * 32 + khf * 8);
      pa.u[1] = *(const U128*)(Pw + lm * 32 + 16 + khf * 8);
#pragma unroll
      for (int nf = 0; nf < 4; ++nf) {
        Frag bv;  // V^T rows: hd contiguous in s -> 16B chunks
        size_t base = ((size_t)bh * HDm + nf * 16 + lm) * Seq + kb0;
        bv.u[0] = *(const U128*)(vth + base + khf * 8);
        bv.u[1] = *(const U128*)(vth + base + 16 + khf * 8);
        o[nf] = wmma_bf16(pa, bv, o[nf]);
      }
      __builtin_amdgcn_wave_barrier();
    }
  }

#pragma unroll
  for (int nf = 0; nf < 4; ++nf) {
#pragma unroll
    for (int r = 0; r < 8; ++r) {
      int token = q0 + r + (khf << 3);
      int col = h * HDm + nf * 16 + lm;
      float val = o[nf][r] / rsum[r];
      oh[((size_t)b * Seq + token) * Dm + col] = f2bf(val);
    }
  }
}

// ---------------- launch ----------------
extern "C" void kernel_launch(void* const* d_in, const int* in_sizes, int n_in,
                              void* d_out, int out_size, void* d_ws, size_t ws_size,
                              hipStream_t stream) {
  (void)in_sizes; (void)n_in; (void)out_size; (void)ws_size;
  const float* x          = (const float*)d_in[0];
  const unsigned char* pm = (const unsigned char*)d_in[1];
  // d_in[2] = attn_mask (recomputed analytically in-kernel)
  const float* in_w  = (const float*)d_in[3];
  const float* in_b  = (const float*)d_in[4];
  const float* out_w = (const float*)d_in[5];
  const float* out_b = (const float*)d_in[6];
  const float* ln1g  = (const float*)d_in[7];
  const float* ln1b  = (const float*)d_in[8];
  const float* ln2g  = (const float*)d_in[9];
  const float* ln2b  = (const float*)d_in[10];
  const float* w1    = (const float*)d_in[11];
  const float* b1    = (const float*)d_in[12];
  const float* w2    = (const float*)d_in[13];
  const float* b2    = (const float*)d_in[14];

  char* base = (char*)d_ws;
  size_t off = 0;
  auto take = [&](size_t bytes) -> char* {
    off = (off + 255) & ~(size_t)255;
    char* p = base + off; off += bytes; return p;
  };
  unsigned short* wqkv = (unsigned short*)take((size_t)3 * Dm * Dm * 2);
  unsigned short* wout = (unsigned short*)take((size_t)Dm * Dm * 2);
  unsigned short* w1h  = (unsigned short*)take((size_t)FFm * Dm * 2);
  unsigned short* w2h  = (unsigned short*)take((size_t)Dm * FFm * 2);
  unsigned short* ah   = (unsigned short*)take((size_t)NTOK * Dm * 2);
  unsigned short* qhp  = (unsigned short*)take((size_t)NTOK * Dm * 2);
  unsigned short* khp  = (unsigned short*)take((size_t)NTOK * Dm * 2);
  unsigned short* vthp = (unsigned short*)take((size_t)NTOK * Dm * 2);
  unsigned short* ohp  = (unsigned short*)take((size_t)NTOK * Dm * 2);
  float*          x1   = (float*)take((size_t)NTOK * Dm * 4);
  unsigned short* fh   = (unsigned short*)take((size_t)NTOK * Dm * 2);
  unsigned short* hh   = (unsigned short*)take((size_t)NTOK * FFm * 2);

  // weight fp32 -> bf16
  cvt_bf16_kernel<<<(3 * Dm * Dm + 255) / 256, 256, 0, stream>>>(in_w, wqkv, 3 * Dm * Dm);
  cvt_bf16_kernel<<<(Dm * Dm + 255) / 256, 256, 0, stream>>>(out_w, wout, Dm * Dm);
  cvt_bf16_kernel<<<(FFm * Dm + 255) / 256, 256, 0, stream>>>(w1, w1h, FFm * Dm);
  cvt_bf16_kernel<<<(Dm * FFm + 255) / 256, 256, 0, stream>>>(w2, w2h, Dm * FFm);

  // LN1 -> bf16 activations
  ln_bf16_kernel<<<NTOK, 256, 0, stream>>>(x, ln1g, ln1b, ah);

  // QKV projection (scatter epilogue)
  GemmParams pq{ah, wqkv, in_b, nullptr, nullptr, nullptr, qhp, khp, vthp, NTOK, 3 * Dm, Dm};
  gemm_bf16_kernel<0><<<dim3((3 * Dm) / 128, NTOK / 128), 256, 0, stream>>>(pq);

  // local+global attention
  attn_kernel<<<(Bsz * Hn * (Seq / 16)) / 4, 128, 0, stream>>>(qhp, khp, vthp, pm, ohp);

  // out-proj + residual -> x1 (fp32)
  GemmParams po{ohp, wout, out_b, x, x1, nullptr, nullptr, nullptr, nullptr, NTOK, Dm, Dm};
  gemm_bf16_kernel<1><<<dim3(Dm / 128, NTOK / 128), 256, 0, stream>>>(po);

  // LN2 -> bf16
  ln_bf16_kernel<<<NTOK, 256, 0, stream>>>(x1, ln2g, ln2b, fh);

  // FF1 + exact GELU -> bf16
  GemmParams p1{fh, w1h, b1, nullptr, nullptr, hh, nullptr, nullptr, nullptr, NTOK, FFm, Dm};
  gemm_bf16_kernel<2><<<dim3(FFm / 128, NTOK / 128), 256, 0, stream>>>(p1);

  // FF2 + residual -> d_out (fp32)
  GemmParams p2{hh, w2h, b2, x1, (float*)d_out, nullptr, nullptr, nullptr, nullptr, NTOK, Dm, FFm};
  gemm_bf16_kernel<1><<<dim3(Dm / 128, NTOK / 128), 256, 0, stream>>>(p2);
}
